// ConvMod_42107859370401
// MI455X (gfx1250) — matrix operand order repaired and checked by reading, not compiled, not run on hardware
//
#include <hip/hip_runtime.h>
#include <hip/hip_bf16.h>

// ---------------------------------------------------------------------------
// Modulated conv (StyleGAN-style) for MI455X / gfx1250, wave32 + WMMA.
//   B=8, H=W=256, CIN=NF=64, K=3, WDIM=512
// Pipeline:
//   1) style_kernel      : style[b][cin] = (w[b] @ fce_kernel)*scale + bias
//   2) modweights_kernel : wppT[b][tap][nf][cin] (fp16, modulated+demodulated)
//   3) conv_wmma_kernel  : implicit GEMM, v_wmma_f32_16x16x32_f16
//      Block = 128 thr (4 waves), tile = 64 pixels x 64 NF.
//      Each wave: 1 NF-group (16) x 4 M-subtiles (4x16 pixels); per
//      (tap, k-chunk) step one B fragment is loaded and reused for 4 WMMAs.
//      72 WMMAs per wave, 288 per block.
// ---------------------------------------------------------------------------

typedef _Float16 v16h __attribute__((ext_vector_type(16)));
typedef _Float16 v8h  __attribute__((ext_vector_type(8)));
typedef float    v8f  __attribute__((ext_vector_type(8)));
typedef float    v4f  __attribute__((ext_vector_type(4)));
typedef _Float16 v4h  __attribute__((ext_vector_type(4)));

#define B_    8
#define H_    256
#define W_    256
#define CIN_  64
#define NF_   64
#define WDIM_ 512

#define MTILE 64                 // pixels per block (4 subtiles of 16)
#define XCOLS (MTILE + 2)        // staged columns incl. halo

// fce_scale = sqrt(1/512); conv_scale = sqrt(1/0.6/576)
#define FCE_SCALE  0.0441941738241592f
#define CONV_SCALE 0.0537913965172995f

// ---------------------------------------------------------------------------
// 1) style[b][c]. 8 blocks x 64 threads.
// ---------------------------------------------------------------------------
__global__ __launch_bounds__(64) void style_kernel(
    const float* __restrict__ w, const float* __restrict__ fk,
    const float* __restrict__ fb, float* __restrict__ style)
{
    const int b = blockIdx.x;
    const int c = threadIdx.x;
    float acc = 0.0f;
    for (int j = 0; j < WDIM_; ++j)               // fk[j*64+c]: coalesced over c
        acc += w[b * WDIM_ + j] * fk[j * CIN_ + c];
    style[b * CIN_ + c] = acc * FCE_SCALE + fb[c];
}

// ---------------------------------------------------------------------------
// 2) Modulate + demodulate, store transposed fp16 weights:
//    wppT[((b*9 + tap)*NF + nf)*CIN + cin].  Grid (NF, B) x 64 threads.
// ---------------------------------------------------------------------------
__global__ __launch_bounds__(64) void modweights_kernel(
    const float* __restrict__ ck, const float* __restrict__ style,
    _Float16* __restrict__ wppT)
{
    const int nf = blockIdx.x;
    const int b  = blockIdx.y;
    const int c  = threadIdx.x;

    const float s = style[b * CIN_ + c] * CONV_SCALE;

    float wv[9];
    float acc = 0.0f;
    #pragma unroll
    for (int t = 0; t < 9; ++t) {
        float v = ck[((size_t)t * CIN_ + c) * NF_ + nf] * s;
        wv[t] = v;
        acc  += v * v;
    }

    __shared__ float red[64];
    red[c] = acc;
    __syncthreads();
    for (int st = 32; st > 0; st >>= 1) {
        if (c < st) red[c] += red[c + st];
        __syncthreads();
    }
    const float wstd = rsqrtf(red[0] + 1e-8f);

    #pragma unroll
    for (int t = 0; t < 9; ++t)
        wppT[(((size_t)b * 9 + t) * NF_ + nf) * CIN_ + c] = (_Float16)(wv[t] * wstd);
}

// ---------------------------------------------------------------------------
// 3) Implicit-GEMM conv via WMMA. Grid: (W/64, H, B), block 128 = 4 waves.
// ---------------------------------------------------------------------------
__global__ __launch_bounds__(128) void conv_wmma_kernel(
    const float* __restrict__ x, const _Float16* __restrict__ wppT,
    const float* __restrict__ cb, float* __restrict__ out)
{
    // x tile staged fp16: 3 rows x 66 cols x 64 cin (SAME-pad zero fill)
    __shared__ __attribute__((aligned(16))) _Float16 xs[3 * XCOLS * CIN_];

    const int w0  = blockIdx.x * MTILE;
    const int h   = blockIdx.y;
    const int b   = blockIdx.z;
    const int tid = threadIdx.x;

    // ---- stage: float4 global loads -> 4-half LDS stores -------------------
    // 3*66*64 halfs = 3168 groups of 4
    for (int q = tid; q < 3 * XCOLS * (CIN_ / 4); q += 128) {
        const int cq   = q & (CIN_ / 4 - 1);      // which 4-channel group
        const int rest = q >> 4;
        const int col  = rest % XCOLS;
        const int ky   = rest / XCOLS;
        const int ih   = h + ky - 1;
        const int iw   = w0 + col - 1;
        v4f v = {};
        if (ih >= 0 && ih < H_ && iw >= 0 && iw < W_)
            v = *(const v4f*)&x[(((size_t)b * H_ + ih) * W_ + iw) * CIN_ + cq * 4];
        v4h hv;
        hv.x = (_Float16)v.x; hv.y = (_Float16)v.y;
        hv.z = (_Float16)v.z; hv.w = (_Float16)v.w;
        *(v4h*)&xs[(size_t)q * 4] = hv;
    }
    __syncthreads();

    const int wave   = tid >> 5;
    const int lane   = tid & 31;
    const int m16    = lane & 15;   // A row (pixel within subtile) / B,C col (nf)
    const int hi     = lane >> 4;   // lane-group selector
    const int nfbase = wave * 16;

    // Per-lane A sub-chunk offsets (contiguous K, matching B):
    //   lanes 0-15 : K {0..7, 8..15};  lanes 16-31: K {16..23, 24..31}
    const int aoff0 = hi ? 16 : 0;
    const int aoff1 = hi ? 24 : 8;

    v8f acc0 = {}, acc1 = {}, acc2 = {}, acc3 = {};

    const _Float16* wb = wppT + (size_t)b * 9 * NF_ * CIN_;

    #pragma unroll
    for (int tap = 0; tap < 9; ++tap) {
        const int ky = tap / 3;
        const int kx = tap % 3;
        // B column for this lane: nf = nfbase + m16, cin contiguous
        const _Float16* wcol = wb + ((size_t)tap * NF_ + nfbase + m16) * CIN_;
        // A base for this lane's pixel row within subtile 0
        const _Float16* arow = &xs[((ky * XCOLS) + (m16 + kx)) * CIN_];

        #pragma unroll
        for (int kk = 0; kk < 2; ++kk) {
            const int kb = kk * 32;

            // B 32x16: lanes 0-15 hold K=kb+0..15, lanes 16-31 K=kb+16..31
            const v16h bf = *(const v16h*)(wcol + kb + hi * 16);

            union { v16h v; v8h h[2]; } a0, a1, a2, a3;
            a0.h[0] = *(const v8h*)(arow + 0 * 16 * CIN_ + kb + aoff0);
            a0.h[1] = *(const v8h*)(arow + 0 * 16 * CIN_ + kb + aoff1);
            a1.h[0] = *(const v8h*)(arow + 1 * 16 * CIN_ + kb + aoff0);
            a1.h[1] = *(const v8h*)(arow + 1 * 16 * CIN_ + kb + aoff1);
            a2.h[0] = *(const v8h*)(arow + 2 * 16 * CIN_ + kb + aoff0);
            a2.h[1] = *(const v8h*)(arow + 2 * 16 * CIN_ + kb + aoff1);
            a3.h[0] = *(const v8h*)(arow + 3 * 16 * CIN_ + kb + aoff0);
            a3.h[1] = *(const v8h*)(arow + 3 * 16 * CIN_ + kb + aoff1);

            acc0 = __builtin_amdgcn_wmma_f32_16x16x32_f16(
                false, a0.v, false, bf, (short)0, acc0, false, false);
            acc1 = __builtin_amdgcn_wmma_f32_16x16x32_f16(
                false, a1.v, false, bf, (short)0, acc1, false, false);
            acc2 = __builtin_amdgcn_wmma_f32_16x16x32_f16(
                false, a2.v, false, bf, (short)0, acc2, false, false);
            acc3 = __builtin_amdgcn_wmma_f32_16x16x32_f16(
                false, a3.v, false, bf, (short)0, acc3, false, false);
        }
    }

    // D 16x16 f32 layout: lane l, VGPR r -> M = r + 8*(l>=16), N = l%16
    const float bias = cb[nfbase + m16];
    const size_t obase0 =
        (((size_t)b * H_ + h) * W_ + w0) * NF_ + nfbase + m16;

    v8f accs[4] = {acc0, acc1, acc2, acc3};
    #pragma unroll
    for (int s = 0; s < 4; ++s) {
        const size_t ob = obase0 + (size_t)s * 16 * NF_;
        #pragma unroll
        for (int r = 0; r < 8; ++r) {
            const int m = r + hi * 8;
            out[ob + (size_t)m * NF_] = accs[s][r] + bias;
        }
    }
}

// ---------------------------------------------------------------------------
// Launcher. Workspace layout:
//   [0, 2048)        : style  (8*64 f32)
//   [4096, 4096+576K): wppT   (8*9*64*64 fp16)
// ---------------------------------------------------------------------------
extern "C" void kernel_launch(void* const* d_in, const int* in_sizes, int n_in,
                              void* d_out, int out_size, void* d_ws, size_t ws_size,
                              hipStream_t stream) {
    const float* x  = (const float*)d_in[0];
    const float* w  = (const float*)d_in[1];
    const float* fk = (const float*)d_in[2];
    const float* fb = (const float*)d_in[3];
    const float* ck = (const float*)d_in[4];
    const float* cb = (const float*)d_in[5];
    float* out = (float*)d_out;

    float*    style = (float*)d_ws;
    _Float16* wppT  = (_Float16*)((char*)d_ws + 4096);

    style_kernel<<<dim3(B_), dim3(64), 0, stream>>>(w, fk, fb, style);
    modweights_kernel<<<dim3(NF_, B_), dim3(64), 0, stream>>>(ck, style, wppT);
    conv_wmma_kernel<<<dim3(W_ / MTILE, H_, B_), dim3(128), 0, stream>>>(x, wppT, cb, out);
}
